// HPCrnn_67542655696940
// MI455X (gfx1250) — matrix operand order repaired and checked by reading, not compiled, not run on hardware
//
#include <hip/hip_runtime.h>
#include <hip/hip_bf16.h>
#include <stddef.h>

// ---------------------------------------------------------------------------
// MI455X (gfx1250) implementation of the hippocampus RNN scan.
//
//  * pack_weights: fp32 -> bf16, rearranged into per-(ntile,kblock) WMMA
//    B-operand lane layout so each lane's fragment is one 32B contiguous load.
//  * ca3drive_kernel: 512x512x512 fp32 GEMM with on-the-fly Gaussian A.
//  * hpc_scan: persistent kernel, 8 WGs x 16 waves, T=512 serial loop;
//    per step two 16x512x512 bf16 WMMA GEMMs per WG, state in LDS.
//    Histories use non-temporal stores (streaming, write-once) so the packed
//    weights + ca3drive stay resident in the 192MB L2 across all 512 steps.
// ---------------------------------------------------------------------------

typedef __bf16 bf16;
typedef bf16  v16bf __attribute__((ext_vector_type(16)));
typedef bf16  v8bf  __attribute__((ext_vector_type(8)));
typedef float v8f   __attribute__((ext_vector_type(8)));

#define BS     128
#define TSTEPS 512
#define NN     512        // ecnum == ca1num == ca3num
#define HIS    (BS * TSTEPS * NN)     // 33554432
#define ACTSZ  (BS * TSTEPS * 2)      // 131072

// Fast sigmoid: v_exp_f32 + v_rcp_f32 (both TRANS32, co-execute with VALU)
// instead of the IEEE v_div_scale/div_fmas sequence.
__device__ __forceinline__ float sigmoidf(float x) {
    return __builtin_amdgcn_rcpf(1.0f + __expf(-x));
}

// Load a 16x32 bf16 A-operand fragment from a row-major [16][512] bf16 LDS
// tile, per the CDNA5 16-bit A-matrix layout:
//   lanes 0-15  : row M=lane,    K = k0+{0..7}  and k0+{16..23}
//   lanes 16-31 : row M=lane-16, K = k0+{8..15} and k0+{24..31}
__device__ __forceinline__ v16bf load_a_frag(const bf16* abase, int lane, int kb) {
    const int m     = lane & 15;
    const int half8 = (lane >> 4) * 8;
    const int k0    = kb * 32;
    const v8bf* p0 = (const v8bf*)(abase + m * NN + k0 + half8);        // 16B aligned
    const v8bf* p1 = (const v8bf*)(abase + m * NN + k0 + 16 + half8);   // 16B aligned
    v8bf lo = *p0;
    v8bf hi = *p1;
    v16bf r;
#pragma unroll
    for (int e = 0; e < 8; ++e) { r[e] = lo[e]; r[e + 8] = hi[e]; }
    return r;
}

// ---------------------------------------------------------------------------
// Pack a row-major fp32 [512][512] weight (W[k][n]) into bf16 WMMA B-operand
// fragment order: frag (nt,kb), lane holds 16 contiguous bf16 (32B aligned):
//   lanes 0-15  : col N=nt*16+lane,    K = kb*32 + {0..7} and {16..23}
//   lanes 16-31 : col N=nt*16+lane-16, K = kb*32 + {8..15} and {24..31}
// Linear packed index = ((nt*16 + kb)*32 + lane)*16 + e.
// ---------------------------------------------------------------------------
__global__ void pack_weights(const float* __restrict__ w, bf16* __restrict__ packed) {
    int idx  = blockIdx.x * blockDim.x + threadIdx.x;   // 0 .. 262143
    int e    = idx & 15;
    int lane = (idx >> 4) & 31;
    int kb   = (idx >> 9) & 15;
    int nt   = (idx >> 13) & 31;
    int K = kb * 32 + ((lane >> 4) * 8) + (e & 7) + ((e >> 3) * 16);
    int N = nt * 16 + (lane & 15);
    packed[idx] = (bf16)w[K * NN + N];
}

// ---------------------------------------------------------------------------
// ca3drive[t][c] = sum_j exp(-(centers[j]-t)^2/(2*sigma^2)) * wca3ca1[j][c]
// centers[j] = linspace(-0.1T, 1.1T, 512)[ca3order[j]]
// ---------------------------------------------------------------------------
__global__ void ca3drive_kernel(const float* __restrict__ wca3ca1,
                                const int*   __restrict__ ca3order,
                                float*       __restrict__ drive) {
    int idx = blockIdx.x * blockDim.x + threadIdx.x;    // t*512 + c
    int t = idx >> 9;
    int c = idx & (NN - 1);
    float x = (float)t;
    float acc = 0.0f;
    const float lo   = -0.1f * (float)TSTEPS;                       // -51.2
    const float step = (1.2f * (float)TSTEPS) / (float)(NN - 1);    // 614.4/511
    for (int j = 0; j < NN; ++j) {
        float center = lo + (float)ca3order[j] * step;
        float d = center - x;
        acc += __expf(-d * d * (1.0f / 50.0f)) * wca3ca1[j * NN + c];
    }
    drive[idx] = acc;
}

// ---------------------------------------------------------------------------
// Persistent scan kernel: blockIdx.x in [0,8), rows r0 = 16*blockIdx.x.
// 512 threads = 16 wave32; wave w owns column tiles {2w, 2w+1}.
// ---------------------------------------------------------------------------
__global__ __launch_bounds__(512, 1) void hpc_scan(
    const float* __restrict__ ec3input,   // [128][512][512]
    const float* __restrict__ ec3_last,   // [128][512]
    const float* __restrict__ ec5_last,   // [128][512]
    const float* __restrict__ ca1bias,    // [512]
    const bf16*  __restrict__ packW1,     // packed wec3ca1
    const bf16*  __restrict__ packW2,     // packed wca1ec5
    const float* __restrict__ ca3drive,   // [512][512]
    const float* __restrict__ wca1act,    // [512][2]
    const float* __restrict__ actbias,    // [2]
    const unsigned char* __restrict__ noise_mask, // [128][512][512]
    float* __restrict__ out)
{
    extern __shared__ char smem[];
    float* ec3_s = (float*)smem;                 // [16][512] fp32   32 KB
    float* ec5_s = ec3_s + 16 * NN;              // [16][512] fp32   32 KB
    bf16*  a_bf  = (bf16*)(ec5_s + 16 * NN);     // [16][512] bf16   16 KB
    bf16*  c_bf  = a_bf + 16 * NN;               // [16][512] bf16   16 KB
    float* red   = (float*)(c_bf + 16 * NN);     // [512]            2 KB

    const int tid  = threadIdx.x;
    const int lane = tid & 31;
    const int wave = tid >> 5;
    const int r0   = blockIdx.x * 16;
    const int nt0  = wave * 2;

    float* out_act = out;                        // [b][t][2]
    float* out_e3h = out + ACTSZ;                // [b][t][n]
    float* out_e5h = out_e3h + HIS;
    float* out_c1h = out_e5h + HIS;
    float* out_e3f = out_c1h + HIS;              // [b][n]
    float* out_e5f = out_e3f + BS * NN;
    float* out_c1f = out_e5f + BS * NN;

    // ---- init recurrent state ----
    for (int i = tid; i < 16 * NN; i += 512) {
        int m = i >> 9, n = i & (NN - 1);
        ec3_s[i] = ec3_last[(r0 + m) * NN + n];
        ec5_s[i] = ec5_last[(r0 + m) * NN + n];
    }
    __syncthreads();

    for (int t = 0; t < TSTEPS; ++t) {
        // (1) stage ec3 as bf16 A operand
        for (int i = tid; i < 16 * NN; i += 512) a_bf[i] = (bf16)ec3_s[i];
        __syncthreads();

        // (2) GEMM1: sig = sigmoid(ec3 @ wec3ca1); ca1 = clip(drive*(1+sig)-bias)
#pragma unroll
        for (int tt = 0; tt < 2; ++tt) {
            const int nt = nt0 + tt;
            v8f acc = {};
#pragma unroll 4
            for (int kb = 0; kb < 16; ++kb) {
                v16bf a = load_a_frag(a_bf, lane, kb);
                const v16bf* pb =
                    (const v16bf*)(packW1 + ((size_t)(nt * 16 + kb) * 32 + lane) * 16);
                __builtin_prefetch(packW1 + ((size_t)(nt * 16 + kb + 1) * 32 + lane) * 16, 0, 1);
                v16bf b = *pb;
                acc = __builtin_amdgcn_wmma_f32_16x16x32_bf16(
                    false, a, false, b, (short)0, acc, false, false);
            }
            const int n  = nt * 16 + (lane & 15);
            const int mb = (lane >> 4) * 8;
            const float drv  = ca3drive[t * NN + n];
            const float bias = ca1bias[n];
#pragma unroll
            for (int r = 0; r < 8; ++r) {
                const int m = mb + r;
                float s  = sigmoidf(acc[r]);
                float c1 = drv * (1.0f + s) - bias;
                c1 = fminf(fmaxf(c1, 0.0f), 1.0f);
                c_bf[m * NN + n] = (bf16)c1;
                __builtin_nontemporal_store(
                    c1, out_c1h + ((size_t)(r0 + m) * TSTEPS + t) * NN + n);
            }
        }
        __syncthreads();

        // (3) action head: act = ca1 @ wca1act + actbias (16x512x2, LDS reduce)
        {
            const int o  = tid >> 4;       // 0..31 output slot: m=o>>1, a=o&1
            const int m  = o >> 1;
            const int a  = o & 1;
            const int k0 = (tid & 15) * 32;
            float p = 0.0f;
            for (int k = k0; k < k0 + 32; ++k)
                p += (float)c_bf[m * NN + k] * wca1act[k * 2 + a];
            red[tid] = p;
        }
        __syncthreads();
        if (tid < 32) {
            float s = actbias[tid & 1];
            for (int i = 0; i < 16; ++i) s += red[tid * 16 + i];
            const int m = tid >> 1, a = tid & 1;
            out_act[((size_t)(r0 + m) * TSTEPS + t) * 2 + a] = s;
        }

        // (4) GEMM2: ec5 += (10*TS)*(ca1 @ wca1ec5); ec5/ec3 pointwise update
#pragma unroll
        for (int tt = 0; tt < 2; ++tt) {
            const int nt = nt0 + tt;
            v8f acc = {};
#pragma unroll 4
            for (int kb = 0; kb < 16; ++kb) {
                v16bf a = load_a_frag(c_bf, lane, kb);
                const v16bf* pb =
                    (const v16bf*)(packW2 + ((size_t)(nt * 16 + kb) * 32 + lane) * 16);
                v16bf b = *pb;
                acc = __builtin_amdgcn_wmma_f32_16x16x32_bf16(
                    false, a, false, b, (short)0, acc, false, false);
            }
            const int n  = nt * 16 + (lane & 15);
            const int mb = (lane >> 4) * 8;
#pragma unroll
            for (int r = 0; r < 8; ++r) {
                const int m = mb + r;
                const size_t gidx = ((size_t)(r0 + m) * TSTEPS + t) * NN + n;
                float e5 = ec5_s[m * NN + n] + acc[r];     // 10*TS == 1.0
                e5 = 0.69f + 0.3f * sigmoidf(4.0f * (e5 - 0.3f));
                float x  = __builtin_nontemporal_load(ec3input + gidx);
                float e3 = ec3_s[m * NN + n];
                e3 = e5 * e3 + 0.6f * (1.0f - e3) * x;
                unsigned char msk = __builtin_nontemporal_load(noise_mask + gidx);
                if (msk) e3 = 0.5f * e3 + 0.3f;
                ec3_s[m * NN + n] = e3;
                ec5_s[m * NN + n] = e5;
                __builtin_nontemporal_store(e3, out_e3h + gidx);
                __builtin_nontemporal_store(e5, out_e5h + gidx);
            }
        }
        __syncthreads();
    }

    // ---- final states ----
    for (int i = tid; i < 16 * NN; i += 512) {
        int m = i >> 9, n = i & (NN - 1);
        out_e3f[(r0 + m) * NN + n] = ec3_s[i];
        out_e5f[(r0 + m) * NN + n] = ec5_s[i];
        out_c1f[(r0 + m) * NN + n] = (float)c_bf[i];
    }
}

// ---------------------------------------------------------------------------
extern "C" void kernel_launch(void* const* d_in, const int* in_sizes, int n_in,
                              void* d_out, int out_size, void* d_ws, size_t ws_size,
                              hipStream_t stream) {
    const float* ec3input = (const float*)d_in[0];
    const float* ec3_last = (const float*)d_in[1];
    const float* ec5_last = (const float*)d_in[2];
    // d_in[3] = ca1_last (unused by the reference computation)
    const float* ca1bias  = (const float*)d_in[4];
    const float* wca3ca1  = (const float*)d_in[5];
    const float* wec3ca1  = (const float*)d_in[6];
    const float* wca1ec5  = (const float*)d_in[7];
    const float* wca1act  = (const float*)d_in[8];
    const float* actbias  = (const float*)d_in[9];
    const unsigned char* noise = (const unsigned char*)d_in[10];
    const int*   ca3order = (const int*)d_in[11];
    float* out = (float*)d_out;

    // workspace: [0,512K) packed wec3ca1 bf16; [512K,1M) packed wca1ec5 bf16;
    //            [1M,2M) ca3drive fp32
    bf16*  packW1 = (bf16*)d_ws;
    bf16*  packW2 = packW1 + NN * NN;
    float* drive  = (float*)((char*)d_ws + 2 * NN * NN * sizeof(bf16));

    pack_weights<<<512, 512, 0, stream>>>(wec3ca1, packW1);
    pack_weights<<<512, 512, 0, stream>>>(wca1ec5, packW2);
    ca3drive_kernel<<<1024, 256, 0, stream>>>(wca3ca1, ca3order, drive);

    const size_t smem_bytes =
        (2 * 16 * NN) * sizeof(float) +   // ec3_s, ec5_s
        (2 * 16 * NN) * sizeof(bf16)  +   // a_bf, c_bf
        512 * sizeof(float);              // red
    hpc_scan<<<8, 512, smem_bytes, stream>>>(
        ec3input, ec3_last, ec5_last, ca1bias, packW1, packW2,
        drive, wca1act, actbias, noise, out);
}
